// InferenceNet_72215580115453
// MI455X (gfx1250) — compile-verified, matrix operand order checked by reading, not run on hardware
//
#include <hip/hip_runtime.h>
#include <hip/hip_bf16.h>
#include <math.h>

// ---------------------------------------------------------------------------
// VRNN inference fused kernel for gfx1250 (MI455X), bf16 WMMA + f32 accum.
// 32 rows per workgroup (2 M-tiles); B fragments software-pipelined so
// global_load_b128 of future fragments overlaps current WMMAs.
// ---------------------------------------------------------------------------

typedef __attribute__((ext_vector_type(16))) __bf16       v16bf;
typedef __attribute__((ext_vector_type(8)))  float        v8f;
typedef __attribute__((ext_vector_type(8)))  unsigned int v8u;

#define BB 4096
#define TT 20

__device__ __forceinline__ unsigned int f2bf(float f) {
  unsigned int u = __builtin_bit_cast(unsigned int, f);
  return (u + 0x7FFFu + ((u >> 16) & 1u)) >> 16;   // RNE f32 -> bf16
}
__device__ __forceinline__ float lrelu(float v) { return v > 0.f ? v : 0.2f * v; }
__device__ __forceinline__ float sigm(float v) { return 1.f / (1.f + expf(-v)); }
__device__ __forceinline__ float softplus(float v) { return v > 20.f ? v : log1pf(expf(v)); }

// A-fragment (16x32 bf16, M striped over lanes) from row-major bf16 LDS buffer.
__device__ __forceinline__ v16bf load_a(const unsigned short* buf, int stride, int k0, int lane) {
  const int m = lane & 15, half = lane >> 4;
  v8u d;
#pragma unroll
  for (int v = 0; v < 8; ++v) {
    const int kb = (v < 4) ? (half * 8 + 2 * v) : (16 + half * 8 + 2 * (v - 4));
    d[v] = *(const unsigned int*)(buf + m * stride + k0 + kb);
  }
  return __builtin_bit_cast(v16bf, d);
}

// B-fragment (32x16 bf16) from pre-packed weights: 1024B/frag, 32B/lane.
__device__ __forceinline__ v16bf load_b(const unsigned int* pk, int frag, int lane) {
  const uint4* p = (const uint4*)(pk + (size_t)frag * 256 + lane * 8);
  uint4 lo = p[0], hi = p[1];
  v8u d = { lo.x, lo.y, lo.z, lo.w, hi.x, hi.y, hi.z, hi.w };
  return __builtin_bit_cast(v16bf, d);
}

__device__ __forceinline__ v8f wmma_bf16(v16bf a, v16bf b, v8f c) {
  return __builtin_amdgcn_wmma_f32_16x16x32_bf16(false, a, false, b, (short)0, c, false, false);
}

// GEMM over one N-tile, two M-tiles sharing B frags; depth-2 B prefetch.
__device__ __forceinline__ void gemm1_m2(const unsigned short* a, int as, int Kt,
                                         const unsigned int* pk, int nt, int lane,
                                         v8f& o0, v8f& o1) {
  v8f c0 = {}, c1 = {};
  const int base = nt * Kt;
  v16bf bq0 = load_b(pk, base, lane);
  v16bf bq1 = (Kt > 1) ? load_b(pk, base + 1, lane) : bq0;
  for (int kt = 0; kt < Kt; ++kt) {
    v16bf bn = bq1;
    if (kt + 2 < Kt) bn = load_b(pk, base + kt + 2, lane);
    c0 = wmma_bf16(load_a(a, as, kt * 32, lane), bq0, c0);
    c1 = wmma_bf16(load_a(a + 16 * as, as, kt * 32, lane), bq0, c1);
    bq0 = bq1; bq1 = bn;
  }
  o0 = c0; o1 = c1;
}

// A = concat of two LDS buffers along K (split at kxt*32); flat contiguous
// fragment indices so the depth-2 prefetch crosses the buffer boundary.
__device__ __forceinline__ void gemm2_m2(const unsigned short* ax, int sx, int kxt,
                                         const unsigned short* ah, int sh, int kht,
                                         const unsigned int* pk, int nt, int lane,
                                         v8f& o0, v8f& o1) {
  v8f c0 = {}, c1 = {};
  const int Kt = kxt + kht;
  const int base = nt * Kt;
  v16bf bq0 = load_b(pk, base, lane);
  v16bf bq1 = load_b(pk, base + 1, lane);
  for (int kt = 0; kt < kxt; ++kt) {
    v16bf bn = bq1;
    if (kt + 2 < Kt) bn = load_b(pk, base + kt + 2, lane);
    c0 = wmma_bf16(load_a(ax, sx, kt * 32, lane), bq0, c0);
    c1 = wmma_bf16(load_a(ax + 16 * sx, sx, kt * 32, lane), bq0, c1);
    bq0 = bq1; bq1 = bn;
  }
  for (int kt = 0; kt < kht; ++kt) {
    v16bf bn = bq1;
    if (kxt + kt + 2 < Kt) bn = load_b(pk, base + kxt + kt + 2, lane);
    c0 = wmma_bf16(load_a(ah, sh, kt * 32, lane), bq0, c0);
    c1 = wmma_bf16(load_a(ah + 16 * sh, sh, kt * 32, lane), bq0, c1);
    bq0 = bq1; bq1 = bn;
  }
  o0 = c0; o1 = c1;
}

// ---------------------------------------------------------------------------
// Weight packer: W[N,K] f32 row-major -> bf16 WMMA B-fragment order.
// frag = nt*(K/32)+kt; per-lane 8 dwords (2 bf16 each) in A-mirror layout.
// ---------------------------------------------------------------------------
__global__ __launch_bounds__(256) void pack_w(const float* __restrict__ W, int K,
                                              unsigned int* __restrict__ dst) {
  const int lane = threadIdx.x & 31;
  const int frag = blockIdx.x * 8 + (threadIdx.x >> 5);
  const int Kt = K >> 5;
  const int nt = frag / Kt, kt = frag - nt * Kt;
  const int n = nt * 16 + (lane & 15);
  const int half = lane >> 4;
  unsigned int* o = dst + (size_t)frag * 256 + lane * 8;
#pragma unroll
  for (int v = 0; v < 8; ++v) {
    const int kb = (v < 4) ? (half * 8 + 2 * v) : (16 + half * 8 + 2 * (v - 4));
    const int k = kt * 32 + kb;
    o[v] = f2bf(W[(size_t)n * K + k]) | (f2bf(W[(size_t)n * K + k + 1]) << 16);
  }
}

// ---------------------------------------------------------------------------
// Fused persistent VRNN kernel: 32 rows per workgroup, full T loop.
// LSTM cell state lives in registers; h round-trips through out_hlo (f32).
// ---------------------------------------------------------------------------
__global__ __launch_bounds__(256)
void vrnn_fused(const float* __restrict__ x,
                const unsigned int* __restrict__ pWm1, const float* __restrict__ bm1,
                const unsigned int* __restrict__ pWm2, const float* __restrict__ bm2,
                const unsigned int* __restrict__ pWv1, const float* __restrict__ bv1,
                const unsigned int* __restrict__ pWv2, const float* __restrict__ bv2,
                const unsigned int* __restrict__ pA,
                const unsigned int* __restrict__ pWphi, const float* __restrict__ bphi,
                const unsigned int* __restrict__ pWih, const unsigned int* __restrict__ pWhh,
                const float* __restrict__ b_ih, const float* __restrict__ b_hh,
                float* __restrict__ out) {
  __shared__ __align__(16) unsigned short s_x[32 * 256];    // x_t tile (bf16)      16KB
  __shared__ __align__(16) unsigned short s_h[32 * 512];    // h (bf16)             32KB
  __shared__ __align__(16) unsigned short s_t1[32 * 512];   // hidden/agg scratch   32KB
  __shared__ __align__(16) unsigned short s_z[32 * 128];    // z = mean (bf16)       8KB
  __shared__ __align__(16) unsigned short s_phi[32 * 1024]; // phi (bf16)           64KB

  const int tid = threadIdx.x, lane = tid & 31, wave = tid >> 5;
  const int b0 = blockIdx.x * 32;
  const int ncol = lane & 15, half = lane >> 4;

  float* out_z    = out;
  float* out_mean = out + (size_t)BB * TT * 128;
  float* out_lv   = out + 2 * ((size_t)BB * TT * 128);
  float* out_ho   = out + 3 * ((size_t)BB * TT * 128);
  float* out_hlo  = out_ho + (size_t)BB * TT * 512;

  // Persistent LSTM cell state: 4 jt-slots x 2 m-tiles x 8 rows per lane.
  float creg[4][2][8];
#pragma unroll
  for (int s = 0; s < 4; ++s)
#pragma unroll
    for (int mt = 0; mt < 2; ++mt)
#pragma unroll
      for (int r = 0; r < 8; ++r) creg[s][mt][r] = 0.f;

  // init h=0 in LDS and h_out[:,0,:]=0 in global
  for (int i = tid; i < 32 * 512; i += 256) {
    s_h[i] = 0;
    int m = i >> 9, j = i & 511;
    out_ho[((size_t)(b0 + m) * TT) * 512 + j] = 0.f;
  }
  __syncthreads();

  for (int t = 0; t < TT; ++t) {
    // stage 0: load x_t (and for t>0, reload h from out_hlo[t-1]) as bf16
    for (int i = tid; i < 32 * 256; i += 256) {
      int m = i >> 8, c = i & 255;
      s_x[i] = (unsigned short)f2bf(x[((size_t)(b0 + m) * TT + t) * 256 + c]);
    }
    if (t > 0) {
      for (int i = tid; i < 32 * 512; i += 256) {
        int m = i >> 9, j = i & 511;
        s_h[i] = (unsigned short)f2bf(out_hlo[((size_t)(b0 + m) * TT + (t - 1)) * 512 + j]);
      }
    }
    __syncthreads();

    // stage 1: mean hidden  [32,768]x[768,512] -> lrelu -> s_t1
#pragma unroll 1
    for (int nt = wave; nt < 32; nt += 8) {
      v8f a0, a1;
      gemm2_m2(s_x, 256, 8, s_h, 512, 16, pWm1, nt, lane, a0, a1);
      const int n = nt * 16 + ncol;
      const float bs = bm1[n];
#pragma unroll
      for (int r = 0; r < 8; ++r) {
        int m = r + half * 8;
        s_t1[m * 512 + n]        = (unsigned short)f2bf(lrelu(a0[r] + bs));
        s_t1[(m + 16) * 512 + n] = (unsigned short)f2bf(lrelu(a1[r] + bs));
      }
    }
    __syncthreads();

    // stage 2: mean = lrelu([32,512]x[512,128]) -> s_z + global z/mean
#pragma unroll 1
    for (int nt = wave; nt < 8; nt += 8) {
      v8f a0, a1;
      gemm1_m2(s_t1, 512, 16, pWm2, nt, lane, a0, a1);
      const int n = nt * 16 + ncol;
      const float bs = bm2[n];
#pragma unroll
      for (int r = 0; r < 8; ++r) {
        int m = r + half * 8;
        float v0 = lrelu(a0[r] + bs), v1 = lrelu(a1[r] + bs);
        s_z[m * 128 + n]        = (unsigned short)f2bf(v0);
        s_z[(m + 16) * 128 + n] = (unsigned short)f2bf(v1);
        size_t o0 = ((size_t)(b0 + m) * TT + t) * 128 + n;
        size_t o1 = ((size_t)(b0 + m + 16) * TT + t) * 128 + n;
        out_z[o0] = v0; out_mean[o0] = v0;
        out_z[o1] = v1; out_mean[o1] = v1;
      }
    }
    __syncthreads();

    // stage 3: logvar hidden -> s_t1 (reuse)
#pragma unroll 1
    for (int nt = wave; nt < 32; nt += 8) {
      v8f a0, a1;
      gemm2_m2(s_x, 256, 8, s_h, 512, 16, pWv1, nt, lane, a0, a1);
      const int n = nt * 16 + ncol;
      const float bs = bv1[n];
#pragma unroll
      for (int r = 0; r < 8; ++r) {
        int m = r + half * 8;
        s_t1[m * 512 + n]        = (unsigned short)f2bf(lrelu(a0[r] + bs));
        s_t1[(m + 16) * 512 + n] = (unsigned short)f2bf(lrelu(a1[r] + bs));
      }
    }
    __syncthreads();

    // stage 4: logvar out (global only)
#pragma unroll 1
    for (int nt = wave; nt < 8; nt += 8) {
      v8f a0, a1;
      gemm1_m2(s_t1, 512, 16, pWv2, nt, lane, a0, a1);
      const int n = nt * 16 + ncol;
      const float bs = bv2[n];
#pragma unroll
      for (int r = 0; r < 8; ++r) {
        int m = r + half * 8;
        out_lv[((size_t)(b0 + m) * TT + t) * 128 + n]      = lrelu(a0[r] + bs);
        out_lv[((size_t)(b0 + m + 16) * TT + t) * 128 + n] = lrelu(a1[r] + bs);
      }
    }
    __syncthreads();

    // stage 5: agg = [x|z] @ A.T  [32,384]x[384,384] -> s_t1 (stride 384, raw)
#pragma unroll 1
    for (int nt = wave; nt < 24; nt += 8) {
      v8f a0, a1;
      gemm2_m2(s_x, 256, 8, s_z, 128, 4, pA, nt, lane, a0, a1);
      const int n = nt * 16 + ncol;
#pragma unroll
      for (int r = 0; r < 8; ++r) {
        int m = r + half * 8;
        s_t1[m * 384 + n]        = (unsigned short)f2bf(a0[r]);
        s_t1[(m + 16) * 384 + n] = (unsigned short)f2bf(a1[r]);
      }
    }
    __syncthreads();

    // stage 6: phi = softplus(lrelu(agg @ Wphi.T + b))  [32,384]x[384,1024]
#pragma unroll 1
    for (int nt = wave; nt < 64; nt += 8) {
      v8f a0, a1;
      gemm1_m2(s_t1, 384, 12, pWphi, nt, lane, a0, a1);
      const int n = nt * 16 + ncol;
      const float bs = bphi[n];
#pragma unroll
      for (int r = 0; r < 8; ++r) {
        int m = r + half * 8;
        s_phi[m * 1024 + n]        = (unsigned short)f2bf(softplus(lrelu(a0[r] + bs)));
        s_phi[(m + 16) * 1024 + n] = (unsigned short)f2bf(softplus(lrelu(a1[r] + bs)));
      }
    }
    __syncthreads();

    // stage 7: LSTM gates + cell update.
    // 4 B frags (i,f,g,o) + 2 A frags feed 8 WMMAs per k-step; next k-step's
    // B set prefetched (incl. across the W_ih -> W_hh boundary).
#pragma unroll
    for (int slot = 0; slot < 4; ++slot) {
      const int jt = wave + slot * 8;
      v8f ai0 = {}, af0 = {}, ag0 = {}, ao0 = {};
      v8f ai1 = {}, af1 = {}, ag1 = {}, ao1 = {};
      v16bf bi = load_b(pWih, (0  + jt) * 32, lane);
      v16bf bf_ = load_b(pWih, (32 + jt) * 32, lane);
      v16bf bg = load_b(pWih, (64 + jt) * 32, lane);
      v16bf bo = load_b(pWih, (96 + jt) * 32, lane);
      for (int kt = 0; kt < 32; ++kt) {  // phi @ W_ih.T, K=1024
        v16bf ni, nf, ng, no;
        if (kt + 1 < 32) {
          ni = load_b(pWih, (0  + jt) * 32 + kt + 1, lane);
          nf = load_b(pWih, (32 + jt) * 32 + kt + 1, lane);
          ng = load_b(pWih, (64 + jt) * 32 + kt + 1, lane);
          no = load_b(pWih, (96 + jt) * 32 + kt + 1, lane);
        } else {                         // prefetch first W_hh fragments
          ni = load_b(pWhh, (0  + jt) * 16, lane);
          nf = load_b(pWhh, (32 + jt) * 16, lane);
          ng = load_b(pWhh, (64 + jt) * 16, lane);
          no = load_b(pWhh, (96 + jt) * 16, lane);
        }
        v16bf a0 = load_a(s_phi, 1024, kt * 32, lane);
        v16bf a1 = load_a(s_phi + 16 * 1024, 1024, kt * 32, lane);
        ai0 = wmma_bf16(a0, bi,  ai0); ai1 = wmma_bf16(a1, bi,  ai1);
        af0 = wmma_bf16(a0, bf_, af0); af1 = wmma_bf16(a1, bf_, af1);
        ag0 = wmma_bf16(a0, bg,  ag0); ag1 = wmma_bf16(a1, bg,  ag1);
        ao0 = wmma_bf16(a0, bo,  ao0); ao1 = wmma_bf16(a1, bo,  ao1);
        bi = ni; bf_ = nf; bg = ng; bo = no;
      }
      for (int kt = 0; kt < 16; ++kt) {  // h @ W_hh.T, K=512
        v16bf ni = bi, nf = bf_, ng = bg, no = bo;
        if (kt + 1 < 16) {
          ni = load_b(pWhh, (0  + jt) * 16 + kt + 1, lane);
          nf = load_b(pWhh, (32 + jt) * 16 + kt + 1, lane);
          ng = load_b(pWhh, (64 + jt) * 16 + kt + 1, lane);
          no = load_b(pWhh, (96 + jt) * 16 + kt + 1, lane);
        }
        v16bf a0 = load_a(s_h, 512, kt * 32, lane);
        v16bf a1 = load_a(s_h + 16 * 512, 512, kt * 32, lane);
        ai0 = wmma_bf16(a0, bi,  ai0); ai1 = wmma_bf16(a1, bi,  ai1);
        af0 = wmma_bf16(a0, bf_, af0); af1 = wmma_bf16(a1, bf_, af1);
        ag0 = wmma_bf16(a0, bg,  ag0); ag1 = wmma_bf16(a1, bg,  ag1);
        ao0 = wmma_bf16(a0, bo,  ao0); ao1 = wmma_bf16(a1, bo,  ao1);
        bi = ni; bf_ = nf; bg = ng; bo = no;
      }
      const int j = jt * 16 + ncol;
      const float bi_s = b_ih[j]        + b_hh[j];
      const float bf_s = b_ih[512 + j]  + b_hh[512 + j];
      const float bg_s = b_ih[1024 + j] + b_hh[1024 + j];
      const float bo_s = b_ih[1536 + j] + b_hh[1536 + j];
#pragma unroll
      for (int r = 0; r < 8; ++r) {
        const int m0 = r + half * 8;
        {
          float cold = creg[slot][0][r];
          float cnew = sigm(af0[r] + bf_s) * cold + sigm(ai0[r] + bi_s) * tanhf(ag0[r] + bg_s);
          float hnew = sigm(ao0[r] + bo_s) * tanhf(cnew);
          creg[slot][0][r] = cnew;
          size_t ob = (size_t)(b0 + m0) * TT;
          out_hlo[(ob + t) * 512 + j] = hnew;
          if (t + 1 < TT) out_ho[(ob + t + 1) * 512 + j] = hnew;
        }
        {
          float cold = creg[slot][1][r];
          float cnew = sigm(af1[r] + bf_s) * cold + sigm(ai1[r] + bi_s) * tanhf(ag1[r] + bg_s);
          float hnew = sigm(ao1[r] + bo_s) * tanhf(cnew);
          creg[slot][1][r] = cnew;
          size_t ob = (size_t)(b0 + m0 + 16) * TT;
          out_hlo[(ob + t) * 512 + j] = hnew;
          if (t + 1 < TT) out_ho[(ob + t + 1) * 512 + j] = hnew;
        }
      }
    }
    __syncthreads();  // all gates reads of s_h/s_phi done; out_hlo visible next iter
  }
}

// ---------------------------------------------------------------------------
extern "C" void kernel_launch(void* const* d_in, const int* in_sizes, int n_in,
                              void* d_out, int out_size, void* d_ws, size_t ws_size,
                              hipStream_t stream) {
  const float* x    = (const float*)d_in[0];
  const float* A    = (const float*)d_in[3];
  const float* Wm1  = (const float*)d_in[4];  const float* bm1 = (const float*)d_in[5];
  const float* Wm2  = (const float*)d_in[6];  const float* bm2 = (const float*)d_in[7];
  const float* Wv1  = (const float*)d_in[8];  const float* bv1 = (const float*)d_in[9];
  const float* Wv2  = (const float*)d_in[10]; const float* bv2 = (const float*)d_in[11];
  const float* Wphi = (const float*)d_in[12]; const float* bphi = (const float*)d_in[13];
  const float* Wih  = (const float*)d_in[14]; const float* Whh  = (const float*)d_in[15];
  const float* bih  = (const float*)d_in[16]; const float* bhh  = (const float*)d_in[17];

  // packed-weight layout in d_ws (fragment index offsets, 256 uints/frag)
  unsigned int* ws = (unsigned int*)d_ws;
  unsigned int* pWm1  = ws + (size_t)0    * 256;  // 768 frags (32x24)
  unsigned int* pWm2  = ws + (size_t)768  * 256;  // 128 frags (8x16)
  unsigned int* pWv1  = ws + (size_t)896  * 256;  // 768
  unsigned int* pWv2  = ws + (size_t)1664 * 256;  // 128
  unsigned int* pA    = ws + (size_t)1792 * 256;  // 288 (24x12)
  unsigned int* pWphi = ws + (size_t)2080 * 256;  // 768 (64x12)
  unsigned int* pWih  = ws + (size_t)2848 * 256;  // 4096 (128x32)
  unsigned int* pWhh  = ws + (size_t)6944 * 256;  // 2048 (128x16)

  pack_w<<<96,  256, 0, stream>>>(Wm1,  768,  pWm1);
  pack_w<<<16,  256, 0, stream>>>(Wm2,  512,  pWm2);
  pack_w<<<96,  256, 0, stream>>>(Wv1,  768,  pWv1);
  pack_w<<<16,  256, 0, stream>>>(Wv2,  512,  pWv2);
  pack_w<<<36,  256, 0, stream>>>(A,    384,  pA);
  pack_w<<<96,  256, 0, stream>>>(Wphi, 384,  pWphi);
  pack_w<<<512, 256, 0, stream>>>(Wih,  1024, pWih);
  pack_w<<<256, 256, 0, stream>>>(Whh,  512,  pWhh);

  vrnn_fused<<<BB / 32, 256, 0, stream>>>(
      x, pWm1, bm1, pWm2, bm2, pWv1, bv1, pWv2, bv2, pA, pWphi, bphi,
      pWih, pWhh, bih, bhh, (float*)d_out);

  (void)in_sizes; (void)n_in; (void)out_size; (void)ws_size;
}
